// GAT_TSC_17609365914388
// MI455X (gfx1250) — matrix operand order repaired and checked by reading, not compiled, not run on hardware
//
#include <hip/hip_runtime.h>
#include <hip/hip_bf16.h>
#include <math.h>

typedef __attribute__((ext_vector_type(16))) __bf16 v16bf;
typedef __attribute__((ext_vector_type(8)))  __bf16 v8bf;
typedef __attribute__((ext_vector_type(8)))  float  v8f;

namespace {
constexpr int   NN     = 10000;
constexpr int   F_IN   = 512;
constexpr int   HID    = 256;
constexpr int   HEADS  = 8;
constexpr int   FHD    = 32;
constexpr int   NCLASS = 40;
constexpr int   MP     = 10016;   // 626*16, padded rows for GEMM A / C tiles
constexpr int   MPZ    = 10048;   // 157*64, padded rows for sim-loss (4 j-tiles per block)
constexpr float TAU    = 0.5f;
constexpr float LAMDA  = 0.5f;
}

__device__ __forceinline__ __bf16 f2bf(float f) {
  unsigned u = __float_as_uint(f);
  unsigned r = u + 0x7FFFu + ((u >> 16) & 1u);   // round-to-nearest-even
  return __builtin_bit_cast(__bf16, (unsigned short)(r >> 16));
}
__device__ __forceinline__ __bf16 bf0() {
  return __builtin_bit_cast(__bf16, (unsigned short)0);
}

// ---------- float -> bf16, 8 elements/thread (vector b128 loads + b128 store), zero row pad ----------
__global__ void GAT_cvt_pad8(const float* __restrict__ src, __bf16* __restrict__ dst,
                             int rows, int cols, int prows) {
  long i = ((long)blockIdx.x * blockDim.x + threadIdx.x) * 8;
  if (i >= (long)prows * cols) return;
  int row = (int)(i / cols);            // cols % 8 == 0 for all users: no row crossing
  v8bf o;
  if (row < rows) {
#pragma unroll
    for (int j = 0; j < 8; j++) o[j] = f2bf(src[i + j]);
  } else {
#pragma unroll
    for (int j = 0; j < 8; j++) o[j] = bf0();
  }
  *(v8bf*)(dst + i) = o;
}

// ---------- convert + pack weights into WMMA B-fragment layout ----------
// Bp[((kt*Ntiles + jt)*32 + lane)*16 + j] = B[(kt*32 + 8*(lane>>4) + (j<8 ? j : j+8))*Ncols + jt*16 + (lane&15)]
__global__ void GAT_packW(const float* __restrict__ Wsrc, __bf16* __restrict__ Bp,
                          int Ncols, int K) {
  int idx = blockIdx.x * blockDim.x + threadIdx.x;
  int total = (K / 32) * (Ncols / 16) * 32;
  if (idx >= total) return;
  int lane = idx & 31, t = idx >> 5;
  int ntiles = Ncols / 16;
  int jt = t % ntiles, kt = t / ntiles;
  int hf = lane >> 4, l16 = lane & 15;
  int col = jt * 16 + l16, kb = kt * 32 + 8 * hf;
  v16bf o;
#pragma unroll
  for (int j = 0; j < 16; j++) o[j] = f2bf(Wsrc[(size_t)(kb + (j < 8 ? j : j + 8)) * Ncols + col]);
  *(v16bf*)(Bp + (size_t)idx * 16) = o;
}

__global__ void GAT_fill(float* __restrict__ p, float v, long n) {
  long i = (long)blockIdx.x * blockDim.x + threadIdx.x;
  if (i < n) p[i] = v;
}

// ---------- dense GEMM: C[MP,Ncols] = A[MP,K](bf16) @ B(packed bf16), f32 acc ----------
// one wave per 16x16 tile; A: two b128 loads/lane; B: one 32B load/lane; 8 plain b32 stores
__global__ void GAT_gemm_bf16(const __bf16* __restrict__ A, const __bf16* __restrict__ Bp,
                              float* __restrict__ C, int Ncols, int K) {
  int it = blockIdx.x, jt = blockIdx.y;
  int ntiles = gridDim.y;
  int lane = threadIdx.x;
  int hf = lane >> 4, l16 = lane & 15;
  const __bf16* arow = A + (size_t)(it * 16 + l16) * K;
  v8f c = {};
  for (int kk = 0; kk < K; kk += 32) {
    __builtin_prefetch(arow + kk + 64, 0, 0);
    int ab = kk + (hf ? 8 : 0);      // A: lanes 0-15 K[0..7,16..23]; lanes 16-31 K[8..15,24..31]
    v8bf a0 = *(const v8bf*)(arow + ab);
    v8bf a1 = *(const v8bf*)(arow + ab + 16);
    v16bf a = __builtin_shufflevector(a0, a1, 0, 1, 2, 3, 4, 5, 6, 7,
                                      8, 9, 10, 11, 12, 13, 14, 15);
    v16bf b = *(const v16bf*)(Bp + ((size_t)((kk >> 5) * ntiles + jt) * 32 + lane) * 16);
    c = __builtin_amdgcn_wmma_f32_16x16x32_bf16(false, a, false, b, (short)0, c, false, false);
  }
  float* cptr = C + (size_t)(it * 16 + 8 * hf) * Ncols + jt * 16 + l16;
#pragma unroll
  for (int r = 0; r < 8; r++) cptr[(size_t)r * Ncols] = c[r];   // C buffer padded to MP rows
}

// ---------- per-node attention scores es/ed ----------
__global__ void GAT_attn(const float* __restrict__ h, const float* __restrict__ a_s,
                         const float* __restrict__ a_d, float* __restrict__ es,
                         float* __restrict__ ed, int fh) {
  int idx = blockIdx.x * blockDim.x + threadIdx.x;
  if (idx >= NN * HEADS) return;
  int node = idx / HEADS, hh = idx % HEADS;
  const float* hp  = h + (size_t)node * HEADS * fh + hh * fh;
  const float* asp = a_s + hh * fh;
  const float* adp = a_d + hh * fh;
  float s = 0.f, d = 0.f;
  for (int j = 0; j < fh; j++) { float v = hp[j]; s += v * asp[j]; d += v * adp[j]; }
  es[idx] = s; ed[idx] = d;
}

__device__ __forceinline__ float lrelu02(float v) { return v < 0.f ? 0.2f * v : v; }

// ---------- edge pass 1: segment max (float atomic-max via int-bits trick) ----------
__global__ void GAT_edge_max(const int* __restrict__ src, const int* __restrict__ dst,
                             const float* __restrict__ es, const float* __restrict__ ed,
                             float* __restrict__ m, int ne) {
  long idx = (long)blockIdx.x * blockDim.x + threadIdx.x;
  if (idx >= (long)ne * HEADS) return;
  int e = (int)(idx / HEADS), hh = (int)(idx % HEADS);
  float v = lrelu02(es[src[e] * HEADS + hh] + ed[dst[e] * HEADS + hh]);
  float* addr = &m[dst[e] * HEADS + hh];
  if (v >= 0.f) atomicMax((int*)addr, __float_as_int(v));
  else          atomicMin((unsigned int*)addr, __float_as_uint(v));
}

// ---------- edge pass 2: segment sum of exp(e - m) ----------
__global__ void GAT_edge_sum(const int* __restrict__ src, const int* __restrict__ dst,
                             const float* __restrict__ es, const float* __restrict__ ed,
                             const float* __restrict__ m, float* __restrict__ z, int ne) {
  long idx = (long)blockIdx.x * blockDim.x + threadIdx.x;
  if (idx >= (long)ne * HEADS) return;
  int e = (int)(idx / HEADS), hh = (int)(idx % HEADS);
  int di = dst[e];
  float v = lrelu02(es[src[e] * HEADS + hh] + ed[di * HEADS + hh]);
  atomicAdd(&z[di * HEADS + hh], __expf(v - m[di * HEADS + hh]));
}

// ---------- edge pass 3: scatter alpha * h[src] into acc[dst] ----------
__global__ void GAT_edge_scatter(const int* __restrict__ src, const int* __restrict__ dst,
                                 const float* __restrict__ es, const float* __restrict__ ed,
                                 const float* __restrict__ m, const float* __restrict__ z,
                                 const float* __restrict__ h, float* __restrict__ acc,
                                 int ne, int fh) {
  long idx = (long)blockIdx.x * blockDim.x + threadIdx.x;
  if (idx >= (long)ne * HEADS) return;
  int e = (int)(idx / HEADS), hh = (int)(idx % HEADS);
  int si = src[e], di = dst[e];
  float v = lrelu02(es[si * HEADS + hh] + ed[di * HEADS + hh]);
  float alpha = __expf(v - m[di * HEADS + hh]) / z[di * HEADS + hh];
  const float* hp = h   + (size_t)si * HEADS * fh + hh * fh;
  float*       ap = acc + (size_t)di * HEADS * fh + hh * fh;
  for (int j = 0; j < fh; j++) atomicAdd(&ap[j], alpha * hp[j]);
}

// ---------- elu + MaskLayer residual mix ----------
__global__ void GAT_elu_mask(const float* __restrict__ acc, const float* __restrict__ last,
                             float beta, float* __restrict__ cur, long n) {
  long i = (long)blockIdx.x * blockDim.x + threadIdx.x;
  if (i >= n) return;
  float v = acc[i];
  v = v > 0.f ? v : (__expf(v) - 1.0f);
  if (beta > 0.f) v = beta * last[i] + (1.0f - beta) * v;
  cur[i] = v;
}

// ---------- row-normalize -> bf16 (zero rows padded out to MPZ) ----------
__global__ void GAT_rownorm(const float* __restrict__ cur, __bf16* __restrict__ zn) {
  int r = blockIdx.x * blockDim.x + threadIdx.x;
  if (r >= MPZ) return;
  __bf16* zp = zn + (size_t)r * HID;
  if (r >= NN) {
    v8bf o;
#pragma unroll
    for (int j = 0; j < 8; j++) o[j] = bf0();
    for (int c = 0; c < HID; c += 8) *(v8bf*)(zp + c) = o;
    return;
  }
  const float* cp = cur + (size_t)r * HID;
  float ss = 0.f;
  for (int c = 0; c < HID; c++) { float v = cp[c]; ss += v * v; }
  float inv = 1.0f / fmaxf(sqrtf(ss), 1e-12f);
  for (int c = 0; c < HID; c += 8) {
    v8bf o;
#pragma unroll
    for (int j = 0; j < 8; j++) o[j] = f2bf(cp[c + j] * inv);
    *(v8bf*)(zp + c) = o;
  }
}

// ---------- fused similarity row-sum: rowsum_i += sum_j exp(zn_i . zn_j / tau) ----------
// 4 waves/block, each wave owns one 16x16 j-tile; the shared 16x256 A-tile (8KB) is staged
// into LDS with async global->LDS b128 copies (ASYNCcnt), fragments read back via ds_load.
__global__ void GAT_sim_loss(const __bf16* __restrict__ zn, float* __restrict__ rowsum,
                             float* __restrict__ dvec) {
  __shared__ __align__(16) __bf16 atile[16 * HID];     // 8 KB
  int it = blockIdx.x;
  int tid = threadIdx.x;
  int wave = tid >> 5, lane = tid & 31;
  int hf = lane >> 4, l16 = lane & 15;

  // --- stage A tile: 8KB = 4 issues x 128 lanes x 16B, per-lane (lds, global) addresses ---
  unsigned ldsbase = (unsigned)(unsigned long long)(uintptr_t)atile;
  unsigned long long gbase =
      (unsigned long long)(uintptr_t)((const char*)zn + (size_t)(it * 16) * HID * sizeof(__bf16));
#pragma unroll
  for (int i = 0; i < 4; i++) {
    int off = (i * 128 + tid) * 16;
    unsigned vl = ldsbase + off;
    unsigned long long ga = gbase + off;
    asm volatile("global_load_async_to_lds_b128 %0, %1, off" :: "v"(vl), "v"(ga) : "memory");
  }
  asm volatile("s_wait_asynccnt 0x0" ::: "memory");
  __syncthreads();

  int jt = blockIdx.y * 4 + wave;                      // j-tile rows < MPZ by construction
  const __bf16* lrow = atile + l16 * HID;
  const __bf16* brow = zn + (size_t)(jt * 16 + l16) * HID;   // B col n == zn row n (zn@zn^T)
  v8f c = {};
#pragma unroll
  for (int kk = 0; kk < HID; kk += 32) {
    int ab = kk + (hf ? 8 : 0);
    v8bf a0 = *(const v8bf*)(lrow + ab);               // ds_load_b128
    v8bf a1 = *(const v8bf*)(lrow + ab + 16);
    v16bf a = __builtin_shufflevector(a0, a1, 0, 1, 2, 3, 4, 5, 6, 7,
                                      8, 9, 10, 11, 12, 13, 14, 15);
    v8bf b0 = *(const v8bf*)(brow + ab);
    v8bf b1 = *(const v8bf*)(brow + ab + 16);
    v16bf b = __builtin_shufflevector(b0, b1, 0, 1, 2, 3, 4, 5, 6, 7,
                                      8, 9, 10, 11, 12, 13, 14, 15);
    c = __builtin_amdgcn_wmma_f32_16x16x32_bf16(false, a, false, b, (short)0, c, false, false);
  }
  int gj = jt * 16 + l16;
#pragma unroll
  for (int r = 0; r < 8; r++) {
    int gi = it * 16 + r + 8 * hf;
    float v = (gi < NN && gj < NN) ? __expf(c[r] * (1.0f / TAU)) : 0.0f;
    if (jt == it && l16 == r + 8 * hf && gi < NN) dvec[gi] = v;   // diagonal element
    float s = v;
#pragma unroll
    for (int mm = 8; mm >= 1; mm >>= 1) s += __shfl_xor(s, mm, 16);
    if (l16 == 0 && gi < NN) atomicAdd(&rowsum[gi], s);
  }
}

// ---------- loss = mean(-log(d / (2*(rowsum - d)))) ----------
__global__ void GAT_loss_reduce(const float* __restrict__ rowsum, const float* __restrict__ dvec,
                                float* __restrict__ outloss) {
  __shared__ float sh[256];
  float acc = 0.f;
  for (int i = threadIdx.x; i < NN; i += 256) {
    float d = dvec[i];
    float off = rowsum[i] - d;
    acc += -__logf(d / (2.0f * off));
  }
  sh[threadIdx.x] = acc;
  __syncthreads();
  for (int s = 128; s > 0; s >>= 1) {
    if ((int)threadIdx.x < s) sh[threadIdx.x] += sh[threadIdx.x + s];
    __syncthreads();
  }
  if (threadIdx.x == 0) outloss[0] = sh[0] / (float)NN;
}

// ---------- mean over heads + log_softmax ----------
__global__ void GAT_mean_logsm(const float* __restrict__ acc, float* __restrict__ out) {
  int n = blockIdx.x * blockDim.x + threadIdx.x;
  if (n >= NN) return;
  float v[NCLASS];
  const float* ap = acc + (size_t)n * HEADS * NCLASS;
#pragma unroll
  for (int c = 0; c < NCLASS; c++) {
    float s = 0.f;
    for (int h = 0; h < HEADS; h++) s += ap[h * NCLASS + c];
    v[c] = s * (1.0f / HEADS);
  }
  float mx = v[0];
#pragma unroll
  for (int c = 1; c < NCLASS; c++) mx = fmaxf(mx, v[c]);
  float se = 0.f;
#pragma unroll
  for (int c = 0; c < NCLASS; c++) se += __expf(v[c] - mx);
  float lse = mx + __logf(se);
#pragma unroll
  for (int c = 0; c < NCLASS; c++) out[(size_t)n * NCLASS + c] = v[c] - lse;
}

extern "C" void kernel_launch(void* const* d_in, const int* in_sizes, int n_in,
                              void* d_out, int out_size, void* d_ws, size_t ws_size,
                              hipStream_t stream) {
  (void)n_in; (void)out_size; (void)ws_size;
  const float* xp   = (const float*)d_in[0];
  const int*   srcp = (const int*)d_in[1];
  const int*   dstp = (const int*)d_in[2];
  const float* Wf[4]  = {(const float*)d_in[3], (const float*)d_in[6],
                         (const float*)d_in[9], (const float*)d_in[12]};
  const float* Asf[4] = {(const float*)d_in[4], (const float*)d_in[7],
                         (const float*)d_in[10], (const float*)d_in[13]};
  const float* Adf[4] = {(const float*)d_in[5], (const float*)d_in[8],
                         (const float*)d_in[11], (const float*)d_in[14]};
  const int ne = in_sizes[1];

  // workspace carve (256B aligned)
  char* w = (char*)d_ws;
  auto carve = [&](size_t bytes) { void* p = (void*)w; w += (bytes + 255) & ~(size_t)255; return p; };
  __bf16* Abf   = (__bf16*)carve((size_t)MP * F_IN * sizeof(__bf16));      // also holds zn (MPZ*HID fits)
  __bf16* Bp    = (__bf16*)carve((size_t)F_IN * HEADS * NCLASS * sizeof(__bf16));
  float*  hbuf  = (float*)carve((size_t)MP * HEADS * NCLASS * sizeof(float));  // padded rows
  float*  esb   = (float*)carve((size_t)NN * HEADS * sizeof(float));
  float*  edb   = (float*)carve((size_t)NN * HEADS * sizeof(float));
  float*  mb    = (float*)carve((size_t)NN * HEADS * sizeof(float));
  float*  zb    = (float*)carve((size_t)NN * HEADS * sizeof(float));
  float*  accb  = (float*)carve((size_t)NN * HEADS * NCLASS * sizeof(float));
  float*  lastb = (float*)carve((size_t)NN * HID * sizeof(float));
  float*  curb  = (float*)carve((size_t)NN * HID * sizeof(float));
  float*  rowsum= (float*)carve((size_t)NN * sizeof(float));
  float*  dvec  = (float*)carve((size_t)NN * sizeof(float));
  __bf16* znbf  = Abf;   // MPZ*HID*2 = 5.1MB <= MP*F_IN*2 = 10.3MB

  auto cdiv = [](long a, long b) { return (int)((a + b - 1) / b); };

  auto run_gat = [&](const float* inp, int fin, const float* Wp, const float* asp,
                     const float* adp, int fh) {
    int HF = HEADS * fh;
    GAT_cvt_pad8<<<cdiv((long)MP * fin / 8, 256), 256, 0, stream>>>(inp, Abf, NN, fin, MP);
    GAT_packW<<<cdiv((long)(fin / 32) * (HF / 16) * 32, 256), 256, 0, stream>>>(Wp, Bp, HF, fin);
    GAT_gemm_bf16<<<dim3(MP / 16, HF / 16), 32, 0, stream>>>(Abf, Bp, hbuf, HF, fin);
    GAT_attn<<<cdiv((long)NN * HEADS, 256), 256, 0, stream>>>(hbuf, asp, adp, esb, edb, fh);
    GAT_fill<<<cdiv((long)NN * HEADS, 256), 256, 0, stream>>>(mb, -INFINITY, (long)NN * HEADS);
    GAT_fill<<<cdiv((long)NN * HEADS, 256), 256, 0, stream>>>(zb, 0.0f, (long)NN * HEADS);
    GAT_edge_max<<<cdiv((long)ne * HEADS, 256), 256, 0, stream>>>(srcp, dstp, esb, edb, mb, ne);
    GAT_edge_sum<<<cdiv((long)ne * HEADS, 256), 256, 0, stream>>>(srcp, dstp, esb, edb, mb, zb, ne);
    GAT_fill<<<cdiv((long)NN * HF, 256), 256, 0, stream>>>(accb, 0.0f, (long)NN * HF);
    GAT_edge_scatter<<<cdiv((long)ne * HEADS, 256), 256, 0, stream>>>(
        srcp, dstp, esb, edb, mb, zb, hbuf, accb, ne, fh);
  };

  const long nh = (long)NN * HID;

  // layer 0: x -> lastb
  run_gat(xp, F_IN, Wf[0], Asf[0], Adf[0], FHD);
  GAT_elu_mask<<<cdiv(nh, 256), 256, 0, stream>>>(accb, nullptr, 0.0f, lastb, nh);
  // layer 1: lastb -> curb (beta = lamda/3)
  run_gat(lastb, HID, Wf[1], Asf[1], Adf[1], FHD);
  GAT_elu_mask<<<cdiv(nh, 256), 256, 0, stream>>>(accb, lastb, LAMDA / 3.0f, curb, nh);
  // layer 2: curb -> lastb (beta = lamda/4)
  run_gat(curb, HID, Wf[2], Asf[2], Adf[2], FHD);
  GAT_elu_mask<<<cdiv(nh, 256), 256, 0, stream>>>(accb, curb, LAMDA / 4.0f, lastb, nh);

  // bind loss (the i==2 term): fused zn@zn^T row-sum with async-LDS staged A tiles
  GAT_rownorm<<<cdiv(MPZ, 128), 128, 0, stream>>>(lastb, znbf);
  GAT_fill<<<cdiv(NN, 256), 256, 0, stream>>>(rowsum, 0.0f, NN);
  GAT_sim_loss<<<dim3(MP / 16, MPZ / 64), 128, 0, stream>>>(znbf, rowsum, dvec);
  GAT_loss_reduce<<<1, 256, 0, stream>>>(rowsum, dvec, (float*)d_out + (size_t)NN * NCLASS);

  // output layer + mean heads + log_softmax
  run_gat(lastb, HID, Wf[3], Asf[3], Adf[3], NCLASS);
  GAT_mean_logsm<<<cdiv(NN, 128), 128, 0, stream>>>(accb, (float*)d_out);
}
